// SNN_Net_49813030699283
// MI455X (gfx1250) — matrix operand order repaired and checked by reading, not compiled, not run on hardware
//
#include <hip/hip_runtime.h>

// ---------------------------------------------------------------------------
// SNU 2-layer spiking net, decomposed into:
//   convert(x,W1,W2 -> bf16)
//   Z1 = X @ W1^T            (bf16 WMMA, f32 accum)   [25600 x 1024]
//   scan1: spikes/mem1 over T (elementwise recurrence), emit Y1 in bf16
//   Z2 = Y1 @ W2^T           (bf16 WMMA, f32 accum)
//   scan2: mem2 over T
// ---------------------------------------------------------------------------

typedef __attribute__((ext_vector_type(16))) __bf16 v16bf;
typedef __attribute__((ext_vector_type(8)))  float  v8f;
typedef unsigned short u16;

#define SNN_T   100
#define SNN_B   256
#define SNN_IN  1024
#define SNN_H   1024
#define L_TAU   0.8f

__device__ __forceinline__ u16 f32_to_bf16_rne(float f) {
    unsigned int u = __builtin_bit_cast(unsigned int, f);
    unsigned int r = u + 0x7FFFu + ((u >> 16) & 1u);
    return (u16)(r >> 16);
}

// ---------------- f32 -> bf16 conversion (grid-stride) ----------------------
__global__ __launch_bounds__(256)
void cvt_f32_bf16(const float* __restrict__ in, u16* __restrict__ out, size_t n) {
    size_t i = (size_t)blockIdx.x * blockDim.x + threadIdx.x;
    size_t stride = (size_t)gridDim.x * blockDim.x;
    for (; i < n; i += stride) out[i] = f32_to_bf16_rne(in[i]);
}

// ---------------- bf16 WMMA GEMM:  C[M,N] = A[M,K] * B[N,K]^T ---------------
// Block: 256 threads = 8 waves (wave32). Block tile 256(M) x 128(N).
// Wave grid 4(M) x 2(N); each wave computes 64x64 = 4x4 fragments of 16x16,
// so every A/B fragment load feeds 4 WMMAs (16 wmma : 16 b128 loads / K-step).
__global__ __launch_bounds__(256)
void gemm_bf16_nt(const u16* __restrict__ A, const u16* __restrict__ B,
                  float* __restrict__ C, int M, int N, int K) {
    const int lane = threadIdx.x & 31;
    const int wave = threadIdx.x >> 5;
    const int wm   = wave >> 1;                 // 0..3
    const int wn   = wave & 1;                  // 0..1
    const int tileM = blockIdx.y * 256 + wm * 64;
    const int tileN = blockIdx.x * 128 + wn * 64;

    // A fragment (16x32 bf16): lanes 0-15 row M=lane, K {0..7,16..23};
    //                          lanes 16-31 same rows, K {8..15,24..31}
    const int arow = lane & 15;
    const int aklo = (lane & 16) ? 8 : 0;       // in halves
    // B fragment (32x16, sourced from row-major W[N,K]): lanes 0-15 col N=lane,
    // K {0..7,16..23}; lanes 16-31 same cols, K {8..15,24..31}
    const int bcol = lane & 15;
    const int bklo = (lane & 16) ? 8 : 0;       // in halves

    v8f acc[4][4];
#pragma unroll
    for (int i = 0; i < 4; ++i)
#pragma unroll
        for (int j = 0; j < 4; ++j) acc[i][j] = (v8f)(0.0f);

    const u16* arow_base[4];
#pragma unroll
    for (int i = 0; i < 4; ++i)
        arow_base[i] = A + (size_t)(tileM + 16 * i + arow) * K + aklo;
    const u16* brow_base[4];
#pragma unroll
    for (int j = 0; j < 4; ++j)
        brow_base[j] = B + (size_t)(tileN + 16 * j + bcol) * K + bklo;

    for (int k0 = 0; k0 < K; k0 += 32) {
        v16bf afrag[4], bfrag[4];
#pragma unroll
        for (int i = 0; i < 4; ++i) {
            const u16* p = arow_base[i] + k0;
            uint4* av = reinterpret_cast<uint4*>(&afrag[i]);
            av[0] = *reinterpret_cast<const uint4*>(p);        // K  +0..7
            av[1] = *reinterpret_cast<const uint4*>(p + 16);   // K +16..23
            __builtin_prefetch(p + 256, 0, 3);                 // ~8 K-steps ahead, WGP scope
        }
#pragma unroll
        for (int j = 0; j < 4; ++j) {
            const u16* p = brow_base[j] + k0;
            uint4* bv = reinterpret_cast<uint4*>(&bfrag[j]);
            bv[0] = *reinterpret_cast<const uint4*>(p);        // K  +0..7
            bv[1] = *reinterpret_cast<const uint4*>(p + 16);   // K +16..23
            __builtin_prefetch(p + 256, 0, 3);
        }
#pragma unroll
        for (int i = 0; i < 4; ++i)
#pragma unroll
            for (int j = 0; j < 4; ++j)
                acc[i][j] = __builtin_amdgcn_wmma_f32_16x16x32_bf16(
                    /*neg_a=*/false, afrag[i], /*neg_b=*/false, bfrag[j],
                    /*c_mod=*/(short)0, acc[i][j],
                    /*reuse_a=*/false, /*reuse_b=*/false);
    }

    // C/D layout: VGPR r -> row r (lanes 0-15) / row 8+r (lanes 16-31), col = lane&15
    const int crow0 = (lane & 16) ? 8 : 0;
    const int ccol  = lane & 15;
#pragma unroll
    for (int i = 0; i < 4; ++i)
#pragma unroll
        for (int j = 0; j < 4; ++j) {
            float* cp = C + (size_t)(tileM + 16 * i + crow0) * N
                          + (tileN + 16 * j + ccol);
#pragma unroll
            for (int r = 0; r < 8; ++r) cp[(size_t)r * N] = acc[i][j][r];
        }
}

// ---------------- layer-1 scan: spikes + mem1 + bf16 spikes -----------------
__global__ __launch_bounds__(256)
void scan_layer1(const float* __restrict__ Z1, const float* __restrict__ b1,
                 float* __restrict__ spikes, float* __restrict__ mem1,
                 u16* __restrict__ y1b, int BH, int Hdim, int T) {
    int idx = blockIdx.x * blockDim.x + threadIdx.x;
    if (idx >= BH) return;
    const float bias = b1[idx % Hdim];
    float s = 0.0f, y = 0.0f;
    for (int t = 0; t < T; ++t) {
        size_t off = (size_t)t * BH + idx;
        float u = Z1[off] + L_TAU * s * (1.0f - y);
        s = fmaxf(u, 0.0f);
        y = (s + bias > 0.0f) ? 1.0f : 0.0f;
        spikes[off] = y;
        mem1[off]   = s;
        y1b[off]    = f32_to_bf16_rne(y);   // exact: y in {0,1}
    }
}

// ---------------- layer-2 scan: mem2 ----------------------------------------
__global__ __launch_bounds__(256)
void scan_layer2(const float* __restrict__ Z2, const float* __restrict__ b2,
                 float* __restrict__ mem2, int BH, int Hdim, int T) {
    int idx = blockIdx.x * blockDim.x + threadIdx.x;
    if (idx >= BH) return;
    const float bias = b2[idx % Hdim];
    float s = 0.0f, y = 0.0f;
    for (int t = 0; t < T; ++t) {
        size_t off = (size_t)t * BH + idx;
        float u = Z2[off] + L_TAU * s * (1.0f - y);
        s = fmaxf(u, 0.0f);
        y = (s + bias > 0.0f) ? 1.0f : 0.0f;
        mem2[off] = s;
    }
}

// ---------------------------------------------------------------------------
extern "C" void kernel_launch(void* const* d_in, const int* in_sizes, int n_in,
                              void* d_out, int out_size, void* d_ws, size_t ws_size,
                              hipStream_t stream) {
    (void)in_sizes; (void)n_in; (void)out_size; (void)ws_size;

    const float* x  = (const float*)d_in[0];   // [T,B,IN]
    const float* W1 = (const float*)d_in[1];   // [H,IN]
    const float* b1 = (const float*)d_in[2];   // [H]
    const float* W2 = (const float*)d_in[3];   // [H,H]
    const float* b2 = (const float*)d_in[4];   // [H]

    const int T = SNN_T, B = SNN_B, IN = SNN_IN, H = SNN_H;
    const size_t TB  = (size_t)T * B;          // GEMM M dimension = 25600
    const size_t TBH = TB * H;
    const size_t BH  = (size_t)B * H;

    // Workspace carve-up (all 16B aligned):
    //   Xb  : T*B*IN bf16      (52.4 MB)
    //   W1b : H*IN bf16        ( 2.1 MB)
    //   W2b : H*H bf16         ( 2.1 MB)
    //   Y1b : T*B*H bf16       (52.4 MB)
    //   Z   : T*B*H f32        (104.9 MB, reused for Z1 then Z2)
    u16* Xb  = (u16*)d_ws;
    u16* W1b = Xb  + (size_t)T * B * IN;
    u16* W2b = W1b + (size_t)H * IN;
    u16* Y1b = W2b + (size_t)H * H;
    float* Z = (float*)(Y1b + TBH);

    float* out_spikes = (float*)d_out;          // [T,B,H]
    float* out_mem1   = out_spikes + TBH;       // [T,B,H]
    float* out_mem2   = out_mem1 + TBH;         // [T,B,H]

    // 1) conversions to bf16
    cvt_f32_bf16<<<2048, 256, 0, stream>>>(x,  Xb,  (size_t)T * B * IN);
    cvt_f32_bf16<<<512,  256, 0, stream>>>(W1, W1b, (size_t)H * IN);
    cvt_f32_bf16<<<512,  256, 0, stream>>>(W2, W2b, (size_t)H * H);

    // 2) Z1 = X @ W1^T    (M=T*B, N=H, K=IN)   block tile 256x128
    dim3 ggrid(H / 128, (unsigned)(TB / 256));
    gemm_bf16_nt<<<ggrid, 256, 0, stream>>>(Xb, W1b, Z, (int)TB, H, IN);

    // 3) layer-1 recurrence over T (parallel across B*H)
    scan_layer1<<<(unsigned)((BH + 255) / 256), 256, 0, stream>>>(
        Z, b1, out_spikes, out_mem1, Y1b, (int)BH, H, T);

    // 4) Z2 = Y1 @ W2^T   (M=T*B, N=H, K=H)   -- Z reused
    gemm_bf16_nt<<<ggrid, 256, 0, stream>>>(Y1b, W2b, Z, (int)TB, H, H);

    // 5) layer-2 recurrence over T
    scan_layer2<<<(unsigned)((BH + 255) / 256), 256, 0, stream>>>(
        Z, b2, out_mem2, (int)BH, H, T);
}
